// Net_74749610819948
// MI455X (gfx1250) — compile-verified, matrix operand order checked by reading, not run on hardware
//
#include <hip/hip_runtime.h>
#include <hip/hip_bf16.h>

typedef __attribute__((ext_vector_type(16))) __bf16 v16bf;
typedef __attribute__((ext_vector_type(8)))  float  v8f;
typedef __attribute__((ext_vector_type(4)))  unsigned int v4u;

#define EPSV 1e-5f

// ---------------------------------------------------------------------------
// Kernel 1: transpose head_w (4096 x 64, f32) -> head_wT (64 x 4096, bf16)
// ---------------------------------------------------------------------------
__global__ __launch_bounds__(256) void convert_headw(const float* __restrict__ hw,
                                                     __bf16* __restrict__ hwT) {
    int idx = blockIdx.x * 256 + threadIdx.x;          // 4096*64 = 262144
    if (idx < 4096 * 64) {
        int kk = idx >> 6;
        int n  = idx & 63;
        hwT[(size_t)n * 4096 + kk] = (__bf16)hw[idx];
    }
}

// ---------------------------------------------------------------------------
// Kernel 2: per-patch encoder.
// One block per (b, p) patch: BN2d, codeword soft-assignment, E aggregation,
// L2 norm -> bf16 En row; pooled path -> shp (with BN1d); cos_x scalar.
// ---------------------------------------------------------------------------
__global__ __launch_bounds__(256) void encode_kernel(
    const float* __restrict__ x,
    const float* __restrict__ g2, const float* __restrict__ b2,
    const float* __restrict__ m2, const float* __restrict__ v2,
    const float* __restrict__ cwg, const float* __restrict__ scale,
    const float* __restrict__ pool_w, const float* __restrict__ pool_b,
    const float* __restrict__ g1, const float* __restrict__ b1,
    const float* __restrict__ m1, const float* __restrict__ v1,
    __bf16* __restrict__ En, float* __restrict__ shp, float* __restrict__ cosx)
{
    __shared__ float Xs[9][512];      // patch values (raw, then BN'd in place)
    __shared__ float cw[8][512];      // codewords
    __shared__ float pools[512];      // raw spatial mean per channel
    __shared__ float ascale[512];
    __shared__ float abias[512];
    __shared__ float dotS[9][8];
    __shared__ float xn2[9];
    __shared__ float cn2[8];
    __shared__ float As[9][8];
    __shared__ float asumS[8];
    __shared__ float rA[8], rB[8], rC[8];
    __shared__ float pred[4][64];

    const int bp  = blockIdx.x;           // 0 .. 18431
    const int b   = bp / 36;
    const int p   = bp % 36;
    const int i0  = p / 6;
    const int j0  = p % 6;
    const int tid = threadIdx.x;
    const int lane = tid & 31;
    const int wave = tid >> 5;

    // ---- load raw 3x3x512 patch -------------------------------------------
    const float* xb = x + (size_t)b * 512 * 64;
    for (int idx = tid; idx < 4608; idx += 256) {
        int n = idx >> 9, d = idx & 511;
        int di = n / 3, dj = n % 3;
        Xs[n][d] = xb[d * 64 + (i0 + di) * 8 + (j0 + dj)];
    }
    __syncthreads();

    // ---- pooled mean, cosine stats, BN coefficients -----------------------
    float lnum = 0.f, lpp = 0.f, lss = 0.f;
    for (int d = tid; d < 512; d += 256) {
        float s = 0.f;
#pragma unroll
        for (int n = 0; n < 9; ++n) s += Xs[n][d];
        float pool = s * (1.0f / 9.0f);
        pools[d] = pool;
        float sub = Xs[4][d];             // center element (1,1)
        lnum += pool * sub;
        lpp  += pool * pool;
        lss  += sub * sub;
        float a = g2[d] * rsqrtf(v2[d] + EPSV);
        ascale[d] = a;
        abias[d]  = b2[d] - m2[d] * a;
    }
#pragma unroll
    for (int off = 16; off; off >>= 1) {
        lnum += __shfl_down(lnum, off, 32);
        lpp  += __shfl_down(lpp,  off, 32);
        lss  += __shfl_down(lss,  off, 32);
    }
    if (lane == 0) { rA[wave] = lnum; rB[wave] = lpp; rC[wave] = lss; }
    __syncthreads();
    if (tid == 0) {
        float n = 0.f, pp = 0.f, ss = 0.f;
        for (int w = 0; w < 8; ++w) { n += rA[w]; pp += rB[w]; ss += rC[w]; }
        float den = fmaxf(sqrtf(pp) * sqrtf(ss), 1e-8f);
        cosx[bp] = (n / den) * 2.0f - 1.0f;   // num/den/0.5 - 1
    }
    __syncthreads();

    // ---- apply BN2d in place; load codewords ------------------------------
    for (int idx = tid; idx < 4608; idx += 256) {
        int n = idx >> 9, d = idx & 511;
        Xs[n][d] = Xs[n][d] * ascale[d] + abias[d];
    }
    for (int idx = tid; idx < 4096; idx += 256)
        cw[idx >> 9][idx & 511] = cwg[idx];
    __syncthreads();

    // ---- 89 length-512 dot products (X.c, X.X, c.c), one per wave-task ----
    for (int t = wave; t < 89; t += 8) {
        const float* va;
        const float* vb;
        if (t < 72)      { va = Xs[t / 8]; vb = cw[t % 8]; }
        else if (t < 81) { va = Xs[t - 72]; vb = Xs[t - 72]; }
        else             { va = cw[t - 81]; vb = cw[t - 81]; }
        float s = 0.f;
        for (int d = lane; d < 512; d += 32) s += va[d] * vb[d];
#pragma unroll
        for (int off = 16; off; off >>= 1) s += __shfl_down(s, off, 32);
        if (lane == 0) {
            if (t < 72)      dotS[t / 8][t % 8] = s;
            else if (t < 81) xn2[t - 72] = s;
            else             cn2[t - 81] = s;
        }
    }
    __syncthreads();

    // ---- softmax over K per position --------------------------------------
    if (tid < 9) {
        int n = tid;
        float e[8], mx = -1e30f;
#pragma unroll
        for (int k = 0; k < 8; ++k) {
            float d2 = xn2[n] + cn2[k] - 2.0f * dotS[n][k];
            e[k] = scale[k] * d2;
            mx = fmaxf(mx, e[k]);
        }
        float s = 0.f;
#pragma unroll
        for (int k = 0; k < 8; ++k) { e[k] = __expf(e[k] - mx); s += e[k]; }
        float inv = 1.0f / s;
#pragma unroll
        for (int k = 0; k < 8; ++k) As[n][k] = e[k] * inv;
    }
    __syncthreads();
    if (tid < 8) {
        float s = 0.f;
#pragma unroll
        for (int n = 0; n < 9; ++n) s += As[n][tid];
        asumS[tid] = s;
    }
    __syncthreads();

    // ---- E aggregation + L2 norm, store bf16 row --------------------------
    float Ev[2][8];
    float ss2 = 0.f;
#pragma unroll
    for (int r = 0; r < 2; ++r) {
        int d = tid + r * 256;
        float xv[9];
#pragma unroll
        for (int n = 0; n < 9; ++n) xv[n] = Xs[n][d];
#pragma unroll
        for (int k = 0; k < 8; ++k) {
            float acc = -asumS[k] * cw[k][d];
#pragma unroll
            for (int n = 0; n < 9; ++n) acc += As[n][k] * xv[n];
            Ev[r][k] = acc;
            ss2 += acc * acc;
        }
    }
#pragma unroll
    for (int off = 16; off; off >>= 1) ss2 += __shfl_down(ss2, off, 32);
    if (lane == 0) rA[wave] = ss2;
    __syncthreads();
    float tot = 0.f;
    for (int w = 0; w < 8; ++w) tot += rA[w];
    float inv = 1.0f / fmaxf(sqrtf(tot), 1e-12f);
    __bf16* Ep = En + (size_t)bp * 4096;
#pragma unroll
    for (int r = 0; r < 2; ++r) {
        int d = tid + r * 256;
#pragma unroll
        for (int k = 0; k < 8; ++k) Ep[k * 512 + d] = (__bf16)(Ev[r][k] * inv);
    }

    // ---- pooled path: p = pools @ pool_w + pool_b; BN1d -> shp ------------
    int f = tid & 63, part = tid >> 6;
    float acc = 0.f;
    for (int d = part * 128; d < (part + 1) * 128; ++d)
        acc += pools[d] * pool_w[d * 64 + f];
    pred[part][f] = acc;
    __syncthreads();
    if (tid < 64) {
        float pv = pred[0][tid] + pred[1][tid] + pred[2][tid] + pred[3][tid] + pool_b[tid];
        float a  = g1[tid] * rsqrtf(v1[tid] + EPSV);
        shp[(size_t)bp * 64 + tid] = (pv - m1[tid]) * a + b1[tid];
    }
}

// ---------------------------------------------------------------------------
// Kernel 3: head GEMM (18432 x 4096) x (4096 x 64) with v_wmma_f32_16x16x32_bf16.
// One wave computes a 16x64 output strip (4 f32 accumulators); 128 K-steps,
// software-pipelined: fragments for step k+1 are loaded before the WMMAs of
// step k so the compiler's s_wait_loadcnt lands after the matrix ops.
// Fragment loads follow the ISA 16-bit layout: lane L holds row M=L%16,
// K-halves {Kb..Kb+7, Kb+16..Kb+23} with Kb = (L<16 ? 0 : 8) -> two b128 loads.
// 2 waves/block -> 576 workgroups to spread across all WGPs.
// ---------------------------------------------------------------------------
union Frag { v16bf v; v4u q[2]; };

__device__ __forceinline__ Frag load_frag(const __bf16* __restrict__ base, int k0) {
    Frag f;
    f.q[0] = *(const v4u*)(base + k0);
    f.q[1] = *(const v4u*)(base + k0 + 16);
    return f;
}

__global__ __launch_bounds__(64) void head_gemm(const __bf16* __restrict__ A,
                                                const __bf16* __restrict__ BT,
                                                const float* __restrict__ bias,
                                                float* __restrict__ C)
{
    const int wave = threadIdx.x >> 5;               // 0..1
    const int lane = threadIdx.x & 31;
    const int rt   = blockIdx.x * 2 + wave;          // 0 .. 1151
    const int row  = rt * 16 + (lane & 15);
    const int kb   = (lane < 16) ? 0 : 8;

    const __bf16* arow  = A  + (size_t)row * 4096 + kb;
    const __bf16* brow0 = BT + (size_t)(0 * 16 + (lane & 15)) * 4096 + kb;
    const __bf16* brow1 = BT + (size_t)(1 * 16 + (lane & 15)) * 4096 + kb;
    const __bf16* brow2 = BT + (size_t)(2 * 16 + (lane & 15)) * 4096 + kb;
    const __bf16* brow3 = BT + (size_t)(3 * 16 + (lane & 15)) * 4096 + kb;

    v8f zero = {0.f, 0.f, 0.f, 0.f, 0.f, 0.f, 0.f, 0.f};
    v8f acc0 = zero, acc1 = zero, acc2 = zero, acc3 = zero;

    // prologue: fragments for k0 = 0
    Frag a_c  = load_frag(arow, 0);
    Frag b0_c = load_frag(brow0, 0);
    Frag b1_c = load_frag(brow1, 0);
    Frag b2_c = load_frag(brow2, 0);
    Frag b3_c = load_frag(brow3, 0);

#pragma unroll 2
    for (int k0 = 0; k0 < 4096; k0 += 32) {
        // issue next step's loads first (last iteration reloads k0 harmlessly)
        const int kn = (k0 + 32 < 4096) ? (k0 + 32) : k0;
        Frag a_n  = load_frag(arow, kn);
        Frag b0_n = load_frag(brow0, kn);
        Frag b1_n = load_frag(brow1, kn);
        Frag b2_n = load_frag(brow2, kn);
        Frag b3_n = load_frag(brow3, kn);

        acc0 = __builtin_amdgcn_wmma_f32_16x16x32_bf16(false, a_c.v, false, b0_c.v,
                                                       (short)0, acc0, false, false);
        acc1 = __builtin_amdgcn_wmma_f32_16x16x32_bf16(false, a_c.v, false, b1_c.v,
                                                       (short)0, acc1, false, false);
        acc2 = __builtin_amdgcn_wmma_f32_16x16x32_bf16(false, a_c.v, false, b2_c.v,
                                                       (short)0, acc2, false, false);
        acc3 = __builtin_amdgcn_wmma_f32_16x16x32_bf16(false, a_c.v, false, b3_c.v,
                                                       (short)0, acc3, false, false);

        a_c = a_n; b0_c = b0_n; b1_c = b1_n; b2_c = b2_n; b3_c = b3_n;
    }

    // C layout per ISA: VGPR r, lanes 0-15 -> M=r; lanes 16-31 -> M=r+8.
    const int rbase = rt * 16 + ((lane >= 16) ? 8 : 0);
    const int col0  = lane & 15;
    v8f accs[4] = {acc0, acc1, acc2, acc3};
#pragma unroll
    for (int nt = 0; nt < 4; ++nt) {
        int col = nt * 16 + col0;
        float bv = bias[col];
#pragma unroll
        for (int r = 0; r < 8; ++r)
            C[(size_t)(rbase + r) * 64 + col] = accs[nt][r] + bv;
    }
}

// ---------------------------------------------------------------------------
// Kernel 4: per-batch attention stack (2x intra + 2x inter), pooling, FC head.
// 36x64 tiles live entirely in LDS; compute is VALU (far below WMMA tile size).
// ---------------------------------------------------------------------------
__device__ __forceinline__ void intra_op(float* X, float* buf, float* S,
                                         const float* __restrict__ W, int tid)
{
    for (int t = tid; t < 1296; t += 256) {
        int n = t / 36, m = t % 36;
        float s = 0.f;
#pragma unroll 8
        for (int f = 0; f < 64; ++f) s += X[n * 64 + f] * X[m * 64 + f];
        S[t] = s * 0.125f;                      // / sqrt(64)
    }
    __syncthreads();
    if (tid < 36) {
        float mx = -1e30f;
        for (int m = 0; m < 36; ++m) mx = fmaxf(mx, S[tid * 36 + m]);
        float ssum = 0.f;
        for (int m = 0; m < 36; ++m) {
            float e = __expf(S[tid * 36 + m] - mx);
            S[tid * 36 + m] = e;
            ssum += e;
        }
        float invs = 1.0f / ssum;
        for (int m = 0; m < 36; ++m) S[tid * 36 + m] *= invs;
    }
    __syncthreads();
    for (int t = tid; t < 2304; t += 256) {
        int n = t >> 6, f = t & 63;
        float s = 0.f;
        for (int m = 0; m < 36; ++m) s += S[n * 36 + m] * X[m * 64 + f];
        buf[t] = s;
    }
    __syncthreads();
    float outv[9];
    int cnt = 0;
    for (int t = tid; t < 2304; t += 256) {
        int n = t >> 6, f = t & 63;
        float s = 0.f;
#pragma unroll 8
        for (int g = 0; g < 64; ++g) s += buf[n * 64 + g] * W[g * 64 + f];
        outv[cnt++] = fmaxf(s, 0.f);
    }
    __syncthreads();
    cnt = 0;
    for (int t = tid; t < 2304; t += 256) X[t] = outv[cnt++];
    __syncthreads();
}

__device__ __forceinline__ void inter_op(float* X, float* Y, float* bufX, float* bufY,
                                         float* S, float* S2,
                                         const float* __restrict__ Wx,
                                         const float* __restrict__ Wy,
                                         const float* cxs, const float* cys, int tid)
{
    for (int t = tid; t < 1296; t += 256) {
        int n = t / 36, m = t % 36;
        float s = 0.f;
#pragma unroll 8
        for (int f = 0; f < 64; ++f) s += X[n * 64 + f] * Y[m * 64 + f];
        S[t] = s * 0.125f;
    }
    __syncthreads();
    // Ayx[m][n] = column softmax of S -> S2
    if (tid < 36) {
        int m = tid;
        float mx = -1e30f;
        for (int n = 0; n < 36; ++n) mx = fmaxf(mx, S[n * 36 + m]);
        float ssum = 0.f;
        for (int n = 0; n < 36; ++n) {
            float e = __expf(S[n * 36 + m] - mx);
            S2[m * 36 + n] = e;
            ssum += e;
        }
        float invs = 1.0f / ssum;
        for (int n = 0; n < 36; ++n) S2[m * 36 + n] *= invs;
    }
    __syncthreads();
    // Axy = row softmax of S, in place
    if (tid < 36) {
        float mx = -1e30f;
        for (int m = 0; m < 36; ++m) mx = fmaxf(mx, S[tid * 36 + m]);
        float ssum = 0.f;
        for (int m = 0; m < 36; ++m) {
            float e = __expf(S[tid * 36 + m] - mx);
            S[tid * 36 + m] = e;
            ssum += e;
        }
        float invs = 1.0f / ssum;
        for (int m = 0; m < 36; ++m) S[tid * 36 + m] *= invs;
    }
    __syncthreads();
    for (int t = tid; t < 2304; t += 256) {
        int n = t >> 6, f = t & 63;
        float sx = 0.f, sy = 0.f;
        for (int m = 0; m < 36; ++m) {
            sx += S[n * 36 + m]  * Y[m * 64 + f];
            sy += S2[n * 36 + m] * X[m * 64 + f];
        }
        bufX[t] = sx;
        bufY[t] = sy;
    }
    __syncthreads();
    float ox[9], oy[9];
    int cnt = 0;
    for (int t = tid; t < 2304; t += 256) {
        int n = t >> 6, f = t & 63;
        float sx = 0.f, sy = 0.f;
#pragma unroll 8
        for (int g = 0; g < 64; ++g) {
            sx += bufX[n * 64 + g] * Wx[g * 64 + f];
            sy += bufY[n * 64 + g] * Wy[g * 64 + f];
        }
        ox[cnt] = X[t] + cxs[n] * sx;
        oy[cnt] = Y[t] + cys[n] * sy;
        ++cnt;
    }
    __syncthreads();
    cnt = 0;
    for (int t = tid; t < 2304; t += 256) { X[t] = ox[cnt]; Y[t] = oy[cnt]; ++cnt; }
    __syncthreads();
}

__global__ __launch_bounds__(256) void attn_kernel(
    const float* __restrict__ texg, const float* __restrict__ shpg,
    const float* __restrict__ cosx,
    const float* __restrict__ Wt1, const float* __restrict__ Ws1,
    const float* __restrict__ Wt2, const float* __restrict__ Ws2,
    const float* __restrict__ i1x, const float* __restrict__ i1y,
    const float* __restrict__ i2x, const float* __restrict__ i2y,
    const float* __restrict__ Wfeat, const float* __restrict__ Wshape,
    const float* __restrict__ fc1w, const float* __restrict__ fc1b,
    const float* __restrict__ fc2w, const float* __restrict__ fc2b,
    float* __restrict__ out)
{
    __shared__ float tX[2304], tY[2304], bufX[2304], bufY[2304];
    __shared__ float S[1296], S2[1296];
    __shared__ float cxs[36], cys[36];
    __shared__ float outc[128], hbuf[128];
    __shared__ float invn, invh;

    const int b   = blockIdx.x;
    const int tid = threadIdx.x;

    for (int t = tid; t < 2304; t += 256) {
        tX[t] = texg[(size_t)b * 2304 + t];
        tY[t] = shpg[(size_t)b * 2304 + t];
    }
    if (tid < 36) {
        float c = cosx[b * 36 + tid];
        cxs[tid] = c;
        cys[tid] = 1.0f - c;
    }
    __syncthreads();

    intra_op(tX, bufX, S, Wt1, tid);
    intra_op(tY, bufX, S, Ws1, tid);
    inter_op(tX, tY, bufX, bufY, S, S2, i1x, i1y, cxs, cys, tid);
    intra_op(tX, bufX, S, Wt2, tid);
    intra_op(tY, bufX, S, Ws2, tid);
    inter_op(tX, tY, bufX, bufY, S, S2, i2x, i2y, cxs, cys, tid);

    // pooling over P with W_feature / W_shape (P x 1)
    if (tid < 128) {
        int f = tid & 63;
        float s = 0.f;
        if (tid < 64) { for (int p = 0; p < 36; ++p) s += tX[p * 64 + f] * Wfeat[p]; }
        else          { for (int p = 0; p < 36; ++p) s += tY[p * 64 + f] * Wshape[p]; }
        outc[tid] = s;
    }
    __syncthreads();
    if (tid == 0) {
        float ss = 0.f;
        for (int i = 0; i < 128; ++i) ss += outc[i] * outc[i];
        invn = 1.0f / fmaxf(sqrtf(ss), 1e-12f);
    }
    __syncthreads();
    if (tid < 128) {
        float s = fc1b[tid];
        for (int i = 0; i < 128; ++i) s += outc[i] * invn * fc1w[i * 128 + tid];
        hbuf[tid] = s;
    }
    __syncthreads();
    if (tid == 0) {
        float ss = 0.f;
        for (int i = 0; i < 128; ++i) ss += hbuf[i] * hbuf[i];
        invh = 1.0f / fmaxf(sqrtf(ss), 1e-12f);
    }
    __syncthreads();
    if (tid < 40) {
        float s = fc2b[tid];
        for (int i = 0; i < 128; ++i) s += hbuf[i] * invh * fc2w[i * 40 + tid];
        out[(size_t)b * 40 + tid] = s;
    }
}

// ---------------------------------------------------------------------------
extern "C" void kernel_launch(void* const* d_in, const int* in_sizes, int n_in,
                              void* d_out, int out_size, void* d_ws, size_t ws_size,
                              hipStream_t stream)
{
    const float* x       = (const float*)d_in[0];
    const float* bn2d_g  = (const float*)d_in[1];
    const float* bn2d_b  = (const float*)d_in[2];
    const float* bn2d_m  = (const float*)d_in[3];
    const float* bn2d_v  = (const float*)d_in[4];
    const float* codew   = (const float*)d_in[5];
    const float* scale   = (const float*)d_in[6];
    const float* head_w  = (const float*)d_in[7];
    const float* head_b  = (const float*)d_in[8];
    const float* pool_w  = (const float*)d_in[9];
    const float* pool_b  = (const float*)d_in[10];
    const float* bn1d_g  = (const float*)d_in[11];
    const float* bn1d_b  = (const float*)d_in[12];
    const float* bn1d_m  = (const float*)d_in[13];
    const float* bn1d_v  = (const float*)d_in[14];
    const float* Wt1     = (const float*)d_in[15];
    const float* Ws1     = (const float*)d_in[16];
    const float* Wt2     = (const float*)d_in[17];
    const float* Ws2     = (const float*)d_in[18];
    const float* i1_Wx   = (const float*)d_in[19];
    const float* i1_Wy   = (const float*)d_in[20];
    const float* i2_Wx   = (const float*)d_in[21];
    const float* i2_Wy   = (const float*)d_in[22];
    const float* W_feat  = (const float*)d_in[23];
    const float* W_shape = (const float*)d_in[24];
    const float* fc1_w   = (const float*)d_in[25];
    const float* fc1_b   = (const float*)d_in[26];
    const float* fc2_w   = (const float*)d_in[27];
    const float* fc2_b   = (const float*)d_in[28];
    float* out = (float*)d_out;

    const size_t BP = 512 * 36;          // 18432 patches

    size_t off = 0;
    auto take = [&](size_t bytes) -> void* {
        void* p = (char*)d_ws + off;
        off += (bytes + 255) & ~(size_t)255;
        return p;
    };
    __bf16* En   = (__bf16*)take(BP * 4096 * sizeof(__bf16));       // ~151 MB
    __bf16* hwT  = (__bf16*)take((size_t)64 * 4096 * sizeof(__bf16));
    float*  texb = (float*)take(BP * 64 * sizeof(float));
    float*  shpb = (float*)take(BP * 64 * sizeof(float));
    float*  cosx = (float*)take(BP * sizeof(float));

    convert_headw<<<(4096 * 64) / 256, 256, 0, stream>>>(head_w, hwT);

    encode_kernel<<<(int)BP, 256, 0, stream>>>(
        x, bn2d_g, bn2d_b, bn2d_m, bn2d_v, codew, scale,
        pool_w, pool_b, bn1d_g, bn1d_b, bn1d_m, bn1d_v,
        En, shpb, cosx);

    // 18432/16 = 1152 row tiles, 2 waves/block -> 576 blocks
    head_gemm<<<576, 64, 0, stream>>>(En, hwT, head_b, texb);

    attn_kernel<<<512, 256, 0, stream>>>(
        texb, shpb, cosx, Wt1, Ws1, Wt2, Ws2,
        i1_Wx, i1_Wy, i2_Wx, i2_Wy, W_feat, W_shape,
        fc1_w, fc1_b, fc2_w, fc2_b, out);
}
